// transformerv1_85194971284001
// MI455X (gfx1250) — matrix operand order repaired and checked by reading, not compiled, not run on hardware
//
#include <hip/hip_runtime.h>
#include <hip/hip_bf16.h>

// ---------------------------------------------------------------------------
// Transformer block forward for MI455X (gfx1250).
// - All matmuls: v_wmma_f32_16x16x32_bf16 (f32 accumulate).
// - Activations/weights converted to bf16 ONCE; GEMM/attention staging is
//   pure bf16 copies -> async global->LDS DMA (ASYNCcnt) where available.
// - Attention: flash-style online softmax, no SxS materialization.
// ---------------------------------------------------------------------------

#define B_ 16
#define S_ 1024
#define D_ 512
#define H_ 8
#define DK_ 64
#define DFF_ 2048
#define M_ (B_ * S_)   // 16384

typedef __attribute__((ext_vector_type(16))) __bf16 v16bf;
typedef __attribute__((ext_vector_type(8)))  float  v8f;

union BFrag {
    v16bf v;
    unsigned short u[16];
    unsigned int   w[8];
    uint4          q[2];
};

__device__ __forceinline__ unsigned short f2bf(float f) {
    unsigned int x = __float_as_uint(f);
    unsigned int r = (x + 0x7FFFu + ((x >> 16) & 1u)) >> 16; // RNE
    return (unsigned short)r;
}

__device__ __forceinline__ v8f wmma_bf16(const v16bf& a, const v16bf& b, const v8f& c) {
    return __builtin_amdgcn_wmma_f32_16x16x32_bf16(
        false, a, false, b, (short)0, c, false, false);
}

// --- async global->LDS 16B copy (CDNA5 GLOBAL_LOAD_ASYNC_TO_LDS_B128) ------
#if __has_builtin(__builtin_amdgcn_global_load_async_to_lds_b128)
#define HAVE_ASYNC 1
typedef int v4i_vs __attribute__((vector_size(16)));
typedef __attribute__((address_space(1))) v4i_vs* as1_v4i;
typedef __attribute__((address_space(3))) v4i_vs* as3_v4i;
__device__ __forceinline__ void cp16_g2l(unsigned short* lds, const unsigned short* g) {
    __builtin_amdgcn_global_load_async_to_lds_b128(
        (as1_v4i)(unsigned short*)g, (as3_v4i)lds, 0, 0);
}
__device__ __forceinline__ void staging_wait() {
#if __has_builtin(__builtin_amdgcn_s_wait_asynccnt)
    __builtin_amdgcn_s_wait_asynccnt(0);
#else
    asm volatile("s_wait_asynccnt 0" ::: "memory");
#endif
}
#else
#define HAVE_ASYNC 0
__device__ __forceinline__ void cp16_g2l(unsigned short* lds, const unsigned short* g) {
    *(uint4*)lds = *(const uint4*)g;
}
__device__ __forceinline__ void staging_wait() {}
#endif

// ---------------------------------------------------------------------------
// one-time converts: f32 -> bf16, and transpose-convert W[z][K][N]->Wt[z][N][K]
// ---------------------------------------------------------------------------
__global__ __launch_bounds__(256)
void cvt_bf16_kernel(const float* __restrict__ in, unsigned short* __restrict__ out,
                     long long n)
{
    const long long i = (long long)blockIdx.x * 256 + threadIdx.x;
    if (i < n) out[i] = f2bf(in[i]);
}

__global__ __launch_bounds__(256)
void transpose_cvt_kernel(const float* __restrict__ W, unsigned short* __restrict__ Wt,
                          int K, int N)
{
    const long long base = (long long)blockIdx.y * K * N;
    const long long idx  = (long long)blockIdx.x * 256 + threadIdx.x;
    if (idx < (long long)K * N) {
        const int n = (int)(idx / K);
        const int k = (int)(idx % K);
        Wt[base + (long long)n * K + k] = f2bf(W[base + (long long)k * N + n]);
    }
}

// ---------------------------------------------------------------------------
// GEMM: C[z] = A (MxK bf16) * Bt[z]^T (Bt is [N][K] bf16) (+bias, opt ReLU)
// Block tile 128x64, 8 waves (4x2), wave tile 32x32, K-step 32.
// Optional f32 and/or bf16 outputs.
// ---------------------------------------------------------------------------
__global__ __launch_bounds__(256)
void gemm_bf16_kernel(const unsigned short* __restrict__ A,
                      const unsigned short* __restrict__ Bt,
                      const float* __restrict__ bias,
                      float* __restrict__ C,            // nullable
                      unsigned short* __restrict__ Cbf, // nullable
                      int M, int N, int K, int relu,
                      long long b_z_stride, long long c_z_stride)
{
    __shared__ __align__(16) unsigned short As[128 * 32]; // [m][k] 8 KB
    __shared__ __align__(16) unsigned short Bs[64 * 32];  // [n][k] 4 KB

    const int tid  = threadIdx.x;
    const int wave = tid >> 5;
    const int lane = tid & 31;
    const int half = lane >> 4;
    const int l16  = lane & 15;
    const int wm   = wave >> 1;   // 0..3
    const int wn   = wave & 1;    // 0..1
    const int bm   = blockIdx.y * 128;
    const int bn   = blockIdx.x * 64;

    const unsigned short* BtZ = Bt + (long long)blockIdx.z * b_z_stride;

    v8f acc[2][2];
#pragma unroll
    for (int mt = 0; mt < 2; ++mt)
#pragma unroll
        for (int nt = 0; nt < 2; ++nt)
#pragma unroll
            for (int r = 0; r < 8; ++r) acc[mt][nt][r] = 0.0f;

    for (int k0 = 0; k0 < K; k0 += 32) {
        // --- stage A tile 128x32 bf16: 512 x 16B chunks, 2 per thread ---
#pragma unroll
        for (int p = 0; p < 2; ++p) {
            const int c   = tid * 2 + p;
            const int row = c >> 2;
            const int off = (c & 3) * 8;
            cp16_g2l(&As[row * 32 + off],
                     A + (long long)(bm + row) * K + k0 + off);
        }
        // --- stage B tile 64x32 bf16: 256 x 16B chunks, 1 per thread ---
        {
            const int n   = tid >> 2;
            const int off = (tid & 3) * 8;
            cp16_g2l(&Bs[n * 32 + off],
                     BtZ + (long long)(bn + n) * K + k0 + off);
        }
        staging_wait();
        __syncthreads();

        // --- fragments (A 16x32 / B 32x16, both in contiguous k order: k = 16*half + s) ---
        BFrag af[2], bfr[2];
#pragma unroll
        for (int mt = 0; mt < 2; ++mt) {
            const int m = wm * 32 + mt * 16 + l16;
            const unsigned int* row = (const unsigned int*)&As[m * 32];
#pragma unroll
            for (int j = 0; j < 8; ++j) {
                const int kb = 2 * j + 16 * half; // even
                af[mt].w[j] = row[kb >> 1];
            }
        }
#pragma unroll
        for (int nt = 0; nt < 2; ++nt) {
            const int n = wn * 32 + nt * 16 + l16;
            const uint4* src = (const uint4*)&Bs[n * 32 + 16 * half];
            bfr[nt].q[0] = src[0];
            bfr[nt].q[1] = src[1];
        }
#pragma unroll
        for (int mt = 0; mt < 2; ++mt)
#pragma unroll
            for (int nt = 0; nt < 2; ++nt)
                acc[mt][nt] = wmma_bf16(af[mt].v, bfr[nt].v, acc[mt][nt]);
        __syncthreads();
    }

    // --- epilogue ---
    float*          Cz  = C   ? C   + (long long)blockIdx.z * c_z_stride : nullptr;
    unsigned short* Cbz = Cbf ? Cbf + (long long)blockIdx.z * c_z_stride : nullptr;
#pragma unroll
    for (int mt = 0; mt < 2; ++mt) {
#pragma unroll
        for (int nt = 0; nt < 2; ++nt) {
            const int n = bn + wn * 32 + nt * 16 + l16;
            const float bv = bias ? bias[n] : 0.0f;
#pragma unroll
            for (int r = 0; r < 8; ++r) {
                const int m = bm + wm * 32 + mt * 16 + r + 8 * half;
                float vv = acc[mt][nt][r] + bv;
                if (relu) vv = fmaxf(vv, 0.0f);
                if (Cz)  Cz[(long long)m * N + n]  = vv;
                if (Cbz) Cbz[(long long)m * N + n] = f2bf(vv);
            }
        }
    }
}

// ---------------------------------------------------------------------------
// Attention: q,k,v bf16 in [H][B*S][64]; out bf16 o[B*S][512] at column h*64.
// grid = (8 q-tiles, 128 b*h), block = 256 (8 waves x 16 queries).
// ---------------------------------------------------------------------------
__global__ __launch_bounds__(256)
void attn_kernel(const unsigned short* __restrict__ q, const unsigned short* __restrict__ k,
                 const unsigned short* __restrict__ v, unsigned short* __restrict__ o)
{
    __shared__ __align__(16) unsigned short Kt[32 * 64];     // [kk][e]  4 KB
    __shared__ __align__(16) unsigned short Vt[64 * 32];     // [e][kk]  4 KB (V^T)
    __shared__ __align__(16) unsigned short Ps[8 * 16 * 32]; // per-wave P 8 KB

    const int tid  = threadIdx.x;
    const int wave = tid >> 5;
    const int lane = tid & 31;
    const int half = lane >> 4;
    const int l16  = lane & 15;

    const int bh = blockIdx.y;
    const int b  = bh / H_;
    const int hi = bh % H_;
    const long long base = (((long long)hi * B_ + b) * S_) * 64;
    const int qrow0 = blockIdx.x * 128 + wave * 16;

    // Q fragments (registers for whole KV loop), A-layout dword gathers
    BFrag qf[2];
    {
        const unsigned int* qp = (const unsigned int*)(q + base + (long long)(qrow0 + l16) * 64);
#pragma unroll
        for (int ec = 0; ec < 2; ++ec)
#pragma unroll
            for (int j = 0; j < 8; ++j) {
                const int kb = 2 * j + 16 * half;
                qf[ec].w[j] = qp[(ec * 32 + kb) >> 1];
            }
    }

    float mrun[8], lrun[8];
    v8f oacc[4];
#pragma unroll
    for (int r = 0; r < 8; ++r) { mrun[r] = -1e30f; lrun[r] = 0.0f; }
#pragma unroll
    for (int et = 0; et < 4; ++et)
#pragma unroll
        for (int r = 0; r < 8; ++r) oacc[et][r] = 0.0f;

    unsigned short* myPs = &Ps[wave * 512];

    for (int jt = 0; jt < 32; ++jt) {
        __syncthreads(); // protect Kt/Vt reuse
        {   // stage K chunk [32 keys][64] bf16: 256 x 16B async chunks
            const int kk = tid >> 3, off = (tid & 7) * 8;
            cp16_g2l(&Kt[kk * 64 + off],
                     k + base + (long long)(jt * 32 + kk) * 64 + off);
        }
        {   // stage V chunk transposed Vt[e][kk] (needs element transpose)
            const int kk = tid & 31, e0 = (tid >> 5) * 8;
            const unsigned short* src = v + base + (long long)(jt * 32 + kk) * 64 + e0;
            uint4 raw = *(const uint4*)src;
            const unsigned short* su = (const unsigned short*)&raw;
#pragma unroll
            for (int i = 0; i < 8; ++i) Vt[(e0 + i) * 32 + kk] = su[i];
        }
        staging_wait();
        __syncthreads();

        // scores tile 16x32 = 2 C-frags
        v8f sc[2];
#pragma unroll
        for (int kn = 0; kn < 2; ++kn) {
            const int kk = kn * 16 + l16;
            BFrag bk0, bk1;
            const uint4* p0 = (const uint4*)&Kt[kk * 64 + 16 * half];
            const uint4* p1 = (const uint4*)&Kt[kk * 64 + 32 + 16 * half];
            bk0.q[0] = p0[0]; bk0.q[1] = p0[1];
            bk1.q[0] = p1[0]; bk1.q[1] = p1[1];
            v8f z;
#pragma unroll
            for (int r = 0; r < 8; ++r) z[r] = 0.0f;
            z = wmma_bf16(qf[0].v, bk0.v, z);
            sc[kn] = wmma_bf16(qf[1].v, bk1.v, z);
        }

        // online softmax per row (scale 1/sqrt(64) applied here)
#pragma unroll
        for (int r = 0; r < 8; ++r) {
            const float s0 = sc[0][r] * 0.125f;
            const float s1 = sc[1][r] * 0.125f;
            float t = fmaxf(s0, s1);
#pragma unroll
            for (int off = 1; off < 16; off <<= 1) t = fmaxf(t, __shfl_xor(t, off, 32));
            const float nm    = fmaxf(mrun[r], t);
            const float alpha = __expf(mrun[r] - nm);
            const float p0 = __expf(s0 - nm);
            const float p1 = __expf(s1 - nm);
            float ts = p0 + p1;
#pragma unroll
            for (int off = 1; off < 16; off <<= 1) ts += __shfl_xor(ts, off, 32);
            lrun[r] = lrun[r] * alpha + ts;
            mrun[r] = nm;
#pragma unroll
            for (int et = 0; et < 4; ++et) oacc[et][r] *= alpha;
            const int m = r + 8 * half;
            myPs[m * 32 + l16]      = f2bf(p0);
            myPs[m * 32 + 16 + l16] = f2bf(p1);
        }
        // same-wave LDS RAW ordering before A-frag reload
        asm volatile("s_wait_dscnt 0" ::: "memory");

        BFrag pf;
        {
            const unsigned int* row = (const unsigned int*)&myPs[l16 * 32];
#pragma unroll
            for (int j = 0; j < 8; ++j) {
                const int kb = 2 * j + 16 * half;
                pf.w[j] = row[kb >> 1];
            }
        }
#pragma unroll
        for (int et = 0; et < 4; ++et) {
            BFrag bv;
            const uint4* p = (const uint4*)&Vt[(et * 16 + l16) * 32 + 16 * half];
            bv.q[0] = p[0]; bv.q[1] = p[1];
            oacc[et] = wmma_bf16(pf.v, bv.v, oacc[et]);
        }
    }

    // write o_bf[B*S][512] at column hi*64
#pragma unroll
    for (int r = 0; r < 8; ++r) {
        const float inv = 1.0f / lrun[r];
        const int s = qrow0 + r + 8 * half;
        unsigned short* dst = o + ((long long)(b * S_ + s)) * 512 + hi * 64;
#pragma unroll
        for (int et = 0; et < 4; ++et) dst[et * 16 + l16] = f2bf(oacc[et][r] * inv);
    }
}

// ---------------------------------------------------------------------------
// Per-batch LayerNorm over (S,D) jointly; apply writes bf16 for the FFN.
// ---------------------------------------------------------------------------
__global__ __launch_bounds__(256)
void ln_stats_kernel(const float* __restrict__ proj, const float* __restrict__ h,
                     float* __restrict__ stats)
{
    __shared__ float ssum[256], ssq[256];
    const int bidx = blockIdx.x;
    const long long n = (long long)S_ * D_;
    const float* p  = proj + bidx * n;
    const float* hh = h    + bidx * n;
    float s = 0.0f, s2 = 0.0f;
    for (long long i = threadIdx.x; i < n; i += 256) {
        const float r = p[i] + hh[i];
        s += r; s2 += r * r;
    }
    ssum[threadIdx.x] = s; ssq[threadIdx.x] = s2;
    __syncthreads();
    for (int st = 128; st > 0; st >>= 1) {
        if (threadIdx.x < st) {
            ssum[threadIdx.x] += ssum[threadIdx.x + st];
            ssq[threadIdx.x]  += ssq[threadIdx.x + st];
        }
        __syncthreads();
    }
    if (threadIdx.x == 0) {
        const float mu  = ssum[0] / (float)n;
        const float var = ssq[0] / (float)n - mu * mu;
        stats[bidx * 2]     = mu;
        stats[bidx * 2 + 1] = rsqrtf(var + 1e-5f);
    }
}

__global__ __launch_bounds__(256)
void ln_apply_kernel(const float* __restrict__ proj, const float* __restrict__ h,
                     const float* __restrict__ stats,
                     const float* __restrict__ lnw, const float* __restrict__ lnb,
                     unsigned short* __restrict__ xn_bf)
{
    const long long i  = (long long)blockIdx.x * 256 + threadIdx.x;
    const long long sd = i % ((long long)S_ * D_);
    const int bidx     = (int)(i / ((long long)S_ * D_));
    const float mu  = stats[bidx * 2];
    const float inv = stats[bidx * 2 + 1];
    xn_bf[i] = f2bf((proj[i] + h[i] - mu) * inv * lnw[sd] + lnb[sd]);
}

// ---------------------------------------------------------------------------
// MaxPool(16) + FC(32->2) + log_softmax. One thread per (b,s).
// ---------------------------------------------------------------------------
__global__ __launch_bounds__(256)
void head_kernel(const float* __restrict__ f, const float* __restrict__ fcw,
                 const float* __restrict__ fcb, float* __restrict__ out)
{
    const int row = blockIdx.x * 256 + threadIdx.x;
    const float* fr = f + (long long)row * 512;
    float z0 = fcb[0], z1 = fcb[1];
#pragma unroll 4
    for (int c = 0; c < 32; ++c) {
        float m = fr[c * 16];
#pragma unroll
        for (int i = 1; i < 16; ++i) m = fmaxf(m, fr[c * 16 + i]);
        z0 += m * fcw[c * 2];
        z1 += m * fcw[c * 2 + 1];
    }
    const float mx  = fmaxf(z0, z1);
    const float lse = mx + logf(__expf(z0 - mx) + __expf(z1 - mx));
    out[row * 2]     = z0 - lse;
    out[row * 2 + 1] = z1 - lse;
}

// ---------------------------------------------------------------------------
extern "C" void kernel_launch(void* const* d_in, const int* in_sizes, int n_in,
                              void* d_out, int out_size, void* d_ws, size_t ws_size,
                              hipStream_t stream)
{
    const float* x     = (const float*)d_in[0];
    const float* emb_w = (const float*)d_in[1];
    const float* emb_b = (const float*)d_in[2];
    const float* WQ    = (const float*)d_in[3];
    const float* WK    = (const float*)d_in[4];
    const float* WV    = (const float*)d_in[5];
    const float* WO    = (const float*)d_in[6];
    const float* ln_w  = (const float*)d_in[7];
    const float* ln_b  = (const float*)d_in[8];
    const float* l1_w  = (const float*)d_in[9];
    const float* l1_b  = (const float*)d_in[10];
    const float* l2_w  = (const float*)d_in[11];
    const float* l2_b  = (const float*)d_in[12];
    const float* fc_w  = (const float*)d_in[13];
    const float* fc_b  = (const float*)d_in[14];

    const size_t MB = 1024ull * 1024ull;
    char* ws = (char*)d_ws;
    // transposed bf16 weights (persist whole run)
    unsigned short* embw_t = (unsigned short*)(ws + 0 * MB);   // [512][512]
    unsigned short* wq_t   = (unsigned short*)(ws + 1 * MB);   // [8][64][512]
    unsigned short* wk_t   = (unsigned short*)(ws + 2 * MB);
    unsigned short* wv_t   = (unsigned short*)(ws + 3 * MB);
    unsigned short* wo_t   = (unsigned short*)(ws + 4 * MB);   // [512][512]
    unsigned short* l1_t   = (unsigned short*)(ws + 5 * MB);   // [2048][512]
    unsigned short* l2_t   = (unsigned short*)(ws + 7 * MB);   // [512][2048]
    // activations
    unsigned short* x_bf   = (unsigned short*)(ws + 16 * MB);  // [M][512] bf16
    float*          h_f32  = (float*)        (ws + 32 * MB);   // [M][512] f32
    unsigned short* h_bf   = (unsigned short*)(ws + 64 * MB);  // [M][512] bf16
    unsigned short* q_bf   = (unsigned short*)(ws + 80 * MB);  // [H][M][64]
    unsigned short* k_bf   = (unsigned short*)(ws + 96 * MB);
    unsigned short* v_bf   = (unsigned short*)(ws + 112 * MB);
    unsigned short* o_bf   = (unsigned short*)(ws + 128 * MB); // [M][512]
    float*          p_f32  = (float*)        (ws + 144 * MB);  // [M][512] proj
    unsigned short* xn_bf  = q_bf;                             // reuse (q dead)
    unsigned short* f1_bf  = k_bf;                             // [M][2048] 96..160MB
    float*          f2_f32 = h_f32;                            // reuse (h dead)
    float*          stats  = (float*)        (ws + 176 * MB);

    const dim3 blk(256);

    // 0) one-time converts
    cvt_bf16_kernel<<<dim3((unsigned)(((long long)M_ * D_ + 255) / 256)), blk, 0, stream>>>(
        x, x_bf, (long long)M_ * D_);
    transpose_cvt_kernel<<<dim3(1024, 1), blk, 0, stream>>>(emb_w, embw_t, D_, D_);
    transpose_cvt_kernel<<<dim3(128, H_), blk, 0, stream>>>(WQ, wq_t, D_, DK_);
    transpose_cvt_kernel<<<dim3(128, H_), blk, 0, stream>>>(WK, wk_t, D_, DK_);
    transpose_cvt_kernel<<<dim3(128, H_), blk, 0, stream>>>(WV, wv_t, D_, DK_);
    transpose_cvt_kernel<<<dim3(1024, 1), blk, 0, stream>>>(WO, wo_t, D_, D_);
    transpose_cvt_kernel<<<dim3(4096, 1), blk, 0, stream>>>(l1_w, l1_t, D_, DFF_);
    transpose_cvt_kernel<<<dim3(4096, 1), blk, 0, stream>>>(l2_w, l2_t, DFF_, D_);

    // 1) embedding: h = x @ emb_w + emb_b  (f32 + bf16 outputs)
    gemm_bf16_kernel<<<dim3(D_ / 64, M_ / 128, 1), blk, 0, stream>>>(
        x_bf, embw_t, emb_b, h_f32, h_bf, M_, D_, D_, 0, 0, 0);

    // 2) QKV projections (grid.z = head), layout [H][M][64], bf16 only
    gemm_bf16_kernel<<<dim3(1, M_ / 128, H_), blk, 0, stream>>>(
        h_bf, wq_t, nullptr, nullptr, q_bf, M_, DK_, D_, 0,
        (long long)DK_ * D_, (long long)M_ * DK_);
    gemm_bf16_kernel<<<dim3(1, M_ / 128, H_), blk, 0, stream>>>(
        h_bf, wk_t, nullptr, nullptr, k_bf, M_, DK_, D_, 0,
        (long long)DK_ * D_, (long long)M_ * DK_);
    gemm_bf16_kernel<<<dim3(1, M_ / 128, H_), blk, 0, stream>>>(
        h_bf, wv_t, nullptr, nullptr, v_bf, M_, DK_, D_, 0,
        (long long)DK_ * D_, (long long)M_ * DK_);

    // 3) attention -> o_bf [M,512]
    attn_kernel<<<dim3(S_ / 128, B_ * H_), blk, 0, stream>>>(q_bf, k_bf, v_bf, o_bf);

    // 4) output projection: proj = o @ WO (f32)
    gemm_bf16_kernel<<<dim3(D_ / 64, M_ / 128, 1), blk, 0, stream>>>(
        o_bf, wo_t, nullptr, p_f32, nullptr, M_, D_, D_, 0, 0, 0);

    // 5) residual + per-batch LayerNorm over (S,D); xn stored bf16
    ln_stats_kernel<<<dim3(B_), blk, 0, stream>>>(p_f32, h_f32, stats);
    ln_apply_kernel<<<dim3((unsigned)((long long)B_ * S_ * D_ / 256)), blk, 0, stream>>>(
        p_f32, h_f32, stats, ln_w, ln_b, xn_bf);

    // 6) FFN: f1 bf16 only (halves traffic), f2 f32 for the head
    gemm_bf16_kernel<<<dim3(DFF_ / 64, M_ / 128, 1), blk, 0, stream>>>(
        xn_bf, l1_t, l1_b, nullptr, f1_bf, M_, DFF_, D_, 1, 0, 0);
    gemm_bf16_kernel<<<dim3(D_ / 64, M_ / 128, 1), blk, 0, stream>>>(
        f1_bf, l2_t, l2_b, f2_f32, nullptr, M_, D_, DFF_, 0, 0, 0);

    // 7) maxpool + fc + log_softmax -> d_out [M,2] f32
    head_kernel<<<dim3(M_ / 256), blk, 0, stream>>>(f2_f32, fc_w, fc_b, (float*)d_out);

    (void)in_sizes; (void)n_in; (void)out_size; (void)ws_size;
}
